// Model_82746839924841
// MI455X (gfx1250) — matrix operand order repaired and checked by reading, not compiled, not run on hardware
//
#include <hip/hip_runtime.h>
#include <hip/hip_bf16.h>
#include <cstdint>

typedef __attribute__((ext_vector_type(16))) __bf16 v16bf;
typedef __attribute__((ext_vector_type(8)))  float  v8f;
typedef __attribute__((ext_vector_type(4)))  int    v4i;
typedef long long ll;
typedef unsigned short u16;

// Problem constants
#define CC   768
#define NN   3136
#define BB   8
#define NHD  12
#define HDIM 64
#define AG   49
#define ROWS (BB*NN)      // 25088
#define BH   (BB*NHD)     // 96

#if __has_builtin(__builtin_amdgcn_global_load_async_to_lds_b128) && \
    __has_builtin(__builtin_amdgcn_s_wait_asynccnt)
#define HAVE_ASYNC_LDS 1
#else
#define HAVE_ASYNC_LDS 0
#endif

typedef __attribute__((address_space(1))) v4i* g4_t;   // global int4*
typedef __attribute__((address_space(3))) v4i* l4_t;   // LDS int4*

__device__ __forceinline__ g4_t to_gas(const void* p) {
    return (g4_t)(uintptr_t)p;
}
__device__ __forceinline__ l4_t to_las(const void* p) {
    return (l4_t)(unsigned)(uintptr_t)p;
}

__device__ __forceinline__ u16 f2bf(float f) {
    unsigned u = __float_as_uint(f);
    unsigned r = (u + 0x7FFFu + ((u >> 16) & 1u)) >> 16;
    return (u16)r;
}

union Frag { uint4 q[2]; v16bf v; };

// ---------------------------------------------------------------------------
// fp32 -> bf16 elementwise
// ---------------------------------------------------------------------------
__global__ void k_f32_to_bf16(const float* __restrict__ in, u16* __restrict__ out, int n) {
    int i = blockIdx.x * 256 + threadIdx.x;
    if (i < n) out[i] = f2bf(in[i]);
}

// ---------------------------------------------------------------------------
// LayerNorm over C for x laid out (B, C, N). One thread per (b, n) row.
// ---------------------------------------------------------------------------
__global__ __launch_bounds__(256) void k_ln_rows(const float* __restrict__ X,
                                                 const float* __restrict__ w,
                                                 const float* __restrict__ bia,
                                                 u16* __restrict__ out) {
    int row = blockIdx.x * 256 + threadIdx.x;
    if (row >= ROWS) return;
    int b = row / NN;
    int n = row - b * NN;
    const float* base = X + (ll)b * CC * NN + n;
    float s = 0.f, ss = 0.f;
    for (int c = 0; c < CC; ++c) {
        float v = base[(ll)c * NN];
        s += v; ss += v * v;
    }
    float mu  = s  * (1.0f / CC);
    float var = ss * (1.0f / CC) - mu * mu;
    float rs  = rsqrtf(var + 1e-5f);
    u16* o = out + (ll)row * CC;
    for (int c = 0; c < CC; ++c) {
        float v = base[(ll)c * NN];
        o[c] = f2bf((v - mu) * rs * w[c] + bia[c]);
    }
}

// Stats only (for y): mean + rsqrt(var+eps) per row
__global__ __launch_bounds__(256) void k_ln_stats(const float* __restrict__ X,
                                                  float* __restrict__ muo,
                                                  float* __restrict__ rso) {
    int row = blockIdx.x * 256 + threadIdx.x;
    if (row >= ROWS) return;
    int b = row / NN;
    int n = row - b * NN;
    const float* base = X + (ll)b * CC * NN + n;
    float s = 0.f, ss = 0.f;
    for (int c = 0; c < CC; ++c) {
        float v = base[(ll)c * NN];
        s += v; ss += v * v;
    }
    float mu  = s  * (1.0f / CC);
    float var = ss * (1.0f / CC) - mu * mu;
    muo[row] = mu;
    rso[row] = rsqrtf(var + 1e-5f);
}

// Pool LN(y) over 8x8 spatial blocks -> 49 agent rows per batch (bf16).
__global__ __launch_bounds__(256) void k_pool_ln(const float* __restrict__ Y,
                                                 const float* __restrict__ mu,
                                                 const float* __restrict__ rs,
                                                 const float* __restrict__ w,
                                                 const float* __restrict__ bia,
                                                 u16* __restrict__ out) {
    int z = blockIdx.x;             // 0..391
    int b = z / AG;
    int a = z - b * AG;
    int ph = a / 7, pw = a - ph * 7;
    int tid = threadIdx.x;
    float acc[3] = {0.f, 0.f, 0.f};
    for (int s = 0; s < 64; ++s) {
        int i = s >> 3, j = s & 7;
        int n = (ph * 8 + i) * 56 + (pw * 8 + j);
        int rowI = b * NN + n;
        float mmu = mu[rowI], rrs = rs[rowI];
        #pragma unroll
        for (int cc = 0; cc < 3; ++cc) {
            int c = tid + cc * 256;
            float v = Y[(ll)b * CC * NN + (ll)c * NN + n];
            acc[cc] += (v - mmu) * rrs;
        }
    }
    #pragma unroll
    for (int cc = 0; cc < 3; ++cc) {
        int c = tid + cc * 256;
        out[(ll)z * CC + c] = f2bf(acc[cc] * (1.0f / 64.0f) * w[c] + bia[c]);
    }
}

// ---------------------------------------------------------------------------
// Generic batched bf16 WMMA GEMM:  C[m,n] = scale * sum_k A(m,k)*B(n,k) (+bias)
// A element (m,k): A[aBase + m*aRow + k]             (k contiguous)
// B element (n,k): B[bBase + n*bN + k*bK]
// C element (m,n): C[cBase + m*cM + n*cN]            (fp32 or bf16)
// Block tile 128(M)x64(N), 8 waves (4Mx2N), wave tile 32x32 (4 wmma/kstep),
// Double-buffered LDS; async-to-LDS DMA overlapped with WMMA of the
// previous tile. Edge tiles use clamped-address loads + select (no EXEC
// branching).
// ---------------------------------------------------------------------------
__global__ __launch_bounds__(256) void k_gemm_bf16(
    const u16* __restrict__ A, const u16* __restrict__ Bw,
    void* __restrict__ Cout, const float* __restrict__ bias,
    int M, int Nn, int K, int div,
    ll aO, ll aI, ll aRow,
    ll bO, ll bI, ll bN, ll bK,
    ll cO, ll cI, ll cM, ll cN,
    float scale, int outBF16) {

    __shared__ u16 As[2][128][32];
    __shared__ u16 Bs[2][64][32];

    int tid  = threadIdx.x;
    int lane = tid & 31;
    int wid  = tid >> 5;
    int waveM = wid & 3;            // 0..3 -> rows 32*waveM
    int waveN = wid >> 2;           // 0..1 -> cols 32*waveN
    int lr   = lane & 15;
    int half = lane >> 4;

    int z  = blockIdx.z;
    int zo = z / div;
    int zi = z - zo * div;
    ll aBase = (ll)zo * aO + (ll)zi * aI;
    ll bBase = (ll)zo * bO + (ll)zi * bI;
    ll cBase = (ll)zo * cO + (ll)zi * cI;

    int mBase = blockIdx.x * 128;
    int nBase = blockIdx.y * 64;

    v8f acc[2][2];
    #pragma unroll
    for (int i = 0; i < 2; ++i)
        #pragma unroll
        for (int j = 0; j < 2; ++j)
            acc[i][j] = (v8f){0.f,0.f,0.f,0.f,0.f,0.f,0.f,0.f};

    // staging assignments
    int arow = tid >> 1;            // 0..127
    int acol = (tid & 1) * 16;      // 0 / 16  (two b128 chunks)
    int brow = tid >> 2;            // 0..63
    int bcol = (tid & 3) * 8;       // one b128 chunk

    const bool aFullMN = (mBase + 128 <= M);
    const bool bFullN  = (nBase + 64 <= Nn);

    // stage one K-step (32) of A and B into buffer `buf`
    auto stage = [&](int buf, int kk) {
        const bool kFull = (kk + 32 <= K);
        // ---- A tile (128 x 32) ----
        if (aFullMN && kFull) {
            const u16* gA = A + aBase + (ll)(mBase + arow) * aRow + kk + acol;
            if (kk + 64 <= K) __builtin_prefetch(gA + 32, 0, 1);
#if HAVE_ASYNC_LDS
            __builtin_amdgcn_global_load_async_to_lds_b128(
                to_gas(gA), to_las(&As[buf][arow][acol]), 0, 0);
            __builtin_amdgcn_global_load_async_to_lds_b128(
                to_gas(gA + 8), to_las(&As[buf][arow][acol + 8]), 0, 0);
#else
            uint4 v0 = *(const uint4*)gA;
            uint4 v1 = *(const uint4*)(gA + 8);
            *(uint4*)&As[buf][arow][acol]     = v0;
            *(uint4*)&As[buf][arow][acol + 8] = v1;
#endif
        } else {
            // clamped addresses + select-to-zero: no EXEC branching
            int gm  = mBase + arow;
            int gmc = gm < M ? gm : (M - 1);
            bool mok = gm < M;
            #pragma unroll
            for (int j = 0; j < 16; ++j) {
                int k  = kk + acol + j;
                int kc = k < K ? k : (K - 1);
                u16 v = A[aBase + (ll)gmc * aRow + kc];
                As[buf][arow][acol + j] = (mok && k < K) ? v : (u16)0;
            }
        }
        // ---- B tile (64 x 32) ----
        const bool bVec = bFullN && kFull && (bK == 1);
        if (bVec) {
            const u16* gB = Bw + bBase + (ll)(nBase + brow) * bN + kk + bcol;
            if (kk + 64 <= K) __builtin_prefetch(gB + 32, 0, 1);
#if HAVE_ASYNC_LDS
            __builtin_amdgcn_global_load_async_to_lds_b128(
                to_gas(gB), to_las(&Bs[buf][brow][bcol]), 0, 0);
#else
            *(uint4*)&Bs[buf][brow][bcol] = *(const uint4*)gB;
#endif
        } else {
            int gn  = nBase + brow;
            int gnc = gn < Nn ? gn : (Nn - 1);
            bool nok = gn < Nn;
            #pragma unroll
            for (int j = 0; j < 8; ++j) {
                int k  = kk + bcol + j;
                int kc = k < K ? k : (K - 1);
                u16 v = Bw[bBase + (ll)gnc * bN + (ll)kc * bK];
                Bs[buf][brow][bcol + j] = (nok && k < K) ? v : (u16)0;
            }
        }
    };

    // consume buffer `buf`: 4 WMMA
    auto compute = [&](int buf) {
        Frag af[2], bfr[2];
        #pragma unroll
        for (int fm = 0; fm < 2; ++fm) {
            const uint4* ar = (const uint4*)(&As[buf][32 * waveM + 16 * fm + lr][0]);
            af[fm].q[0] = ar[half];
            af[fm].q[1] = ar[2 + half];
        }
        #pragma unroll
        for (int fn = 0; fn < 2; ++fn) {
            const uint4* br = (const uint4*)(&Bs[buf][32 * waveN + 16 * fn + lr][0]);
            bfr[fn].q[0] = br[half];
            bfr[fn].q[1] = br[2 + half];
        }
        #pragma unroll
        for (int fm = 0; fm < 2; ++fm)
            #pragma unroll
            for (int fn = 0; fn < 2; ++fn)
                acc[fm][fn] = __builtin_amdgcn_wmma_f32_16x16x32_bf16(
                    false, af[fm].v, false, bfr[fn].v, (short)0, acc[fm][fn], false, false);
    };

    // prologue: stage first tile
    stage(0, 0);
#if HAVE_ASYNC_LDS
    __builtin_amdgcn_s_wait_asynccnt(0);
#endif
    __syncthreads();

    int buf = 0;
    for (int kk = 32; kk < K; kk += 32) {
        stage(buf ^ 1, kk);     // async DMA of next tile...
        compute(buf);           // ...overlapped with WMMA of current tile
#if HAVE_ASYNC_LDS
        __builtin_amdgcn_s_wait_asynccnt(0);
#endif
        __syncthreads();
        buf ^= 1;
    }
    compute(buf);

    // ---- epilogue ----
    if (aFullMN && bFullN) {
        #pragma unroll
        for (int fm = 0; fm < 2; ++fm)
            #pragma unroll
            for (int fn = 0; fn < 2; ++fn)
                #pragma unroll
                for (int r = 0; r < 8; ++r) {
                    int gm = mBase + 32 * waveM + 16 * fm + r + 8 * half;
                    int gn = nBase + 32 * waveN + 16 * fn + lr;
                    float v = acc[fm][fn][r] * scale;
                    if (bias) v += bias[gn];
                    ll off = cBase + (ll)gm * cM + (ll)gn * cN;
                    if (outBF16) ((u16*)Cout)[off] = f2bf(v);
                    else         ((float*)Cout)[off] = v;
                }
    } else {
        #pragma unroll
        for (int fm = 0; fm < 2; ++fm)
            #pragma unroll
            for (int fn = 0; fn < 2; ++fn)
                #pragma unroll
                for (int r = 0; r < 8; ++r) {
                    int gm = mBase + 32 * waveM + 16 * fm + r + 8 * half;
                    int gn = nBase + 32 * waveN + 16 * fn + lr;
                    if (gm < M && gn < Nn) {
                        float v = acc[fm][fn][r] * scale;
                        if (bias) v += bias[gn];
                        ll off = cBase + (ll)gm * cM + (ll)gn * cN;
                        if (outBF16) ((u16*)Cout)[off] = f2bf(v);
                        else         ((float*)Cout)[off] = v;
                    }
                }
    }
}

// ---------------------------------------------------------------------------
// Row softmax over 3136 (agent attention rows), fp32 in -> bf16 out.
// ---------------------------------------------------------------------------
__global__ __launch_bounds__(256) void k_softmax3136(const float* __restrict__ S,
                                                     u16* __restrict__ Aout) {
    __shared__ float red[256];
    int row = blockIdx.x;
    int tid = threadIdx.x;
    const float* p = S + (ll)row * NN;
    float e[13];
    float m = -1e30f;
    #pragma unroll
    for (int j = 0; j < 13; ++j) {
        int i = tid + j * 256;
        if (i < NN) m = fmaxf(m, p[i]);
    }
    red[tid] = m; __syncthreads();
    for (int s = 128; s > 0; s >>= 1) {
        if (tid < s) red[tid] = fmaxf(red[tid], red[tid + s]);
        __syncthreads();
    }
    m = red[0]; __syncthreads();
    float sum = 0.f;
    #pragma unroll
    for (int j = 0; j < 13; ++j) {
        int i = tid + j * 256;
        if (i < NN) { e[j] = __expf(p[i] - m); sum += e[j]; }
        else        e[j] = 0.f;
    }
    red[tid] = sum; __syncthreads();
    for (int s = 128; s > 0; s >>= 1) {
        if (tid < s) red[tid] += red[tid + s];
        __syncthreads();
    }
    float inv = 1.0f / red[0];
    #pragma unroll
    for (int j = 0; j < 13; ++j) {
        int i = tid + j * 256;
        if (i < NN) Aout[(ll)row * NN + i] = f2bf(e[j] * inv);
    }
}

// ---------------------------------------------------------------------------
// Fused q_attn: per (b,h), 64-row tile: S = 0.125 * q @ at^T (WMMA, K=64),
// in-register softmax over 49 agents, bf16 store (padded cols exact zero).
// Padded agent rows are loaded clamped (junk) and forced to -inf pre-softmax.
// ---------------------------------------------------------------------------
__global__ __launch_bounds__(128) void k_qattn(const u16* __restrict__ qb,
                                               const u16* __restrict__ atb,
                                               u16* __restrict__ attn2) {
    int tid  = threadIdx.x;
    int lane = tid & 31;
    int w    = tid >> 5;
    int lr   = lane & 15;
    int half = lane >> 4;
    int z = blockIdx.y;           // b*12 + h
    int b = z / NHD;
    int h = z - b * NHD;
    int rowBase = blockIdx.x * 64 + w * 16;

    const u16* qrow = qb + ((ll)b * NN + rowBase + lr) * CC + h * HDIM;

    v8f acc[4];
    #pragma unroll
    for (int f = 0; f < 4; ++f)
        acc[f] = (v8f){0.f,0.f,0.f,0.f,0.f,0.f,0.f,0.f};

    #pragma unroll
    for (int kk = 0; kk < 64; kk += 32) {
        Frag a;
        a.q[0] = *(const uint4*)(qrow + kk + 8 * half);
        a.q[1] = *(const uint4*)(qrow + kk + 16 + 8 * half);
        #pragma unroll
        for (int f = 0; f < 4; ++f) {
            int ai  = 16 * f + lr;
            int aic = ai < AG ? ai : (AG - 1);     // clamp; masked later
            const u16* arow = atb + ((ll)b * AG + aic) * CC + h * HDIM;
            Frag bb;
            bb.q[0] = *(const uint4*)(arow + kk + 8 * half);
            bb.q[1] = *(const uint4*)(arow + kk + 16 + 8 * half);
            acc[f] = __builtin_amdgcn_wmma_f32_16x16x32_bf16(false, a.v, false, bb.v,
                                                             (short)0, acc[f], false, false);
        }
    }

    float s[4][8];
    #pragma unroll
    for (int f = 0; f < 4; ++f) {
        int ai = 16 * f + lr;
        #pragma unroll
        for (int r = 0; r < 8; ++r) {
            float v = acc[f][r] * 0.125f;
            if (ai >= AG) v = -1e30f;
            s[f][r] = v;
        }
    }

    #pragma unroll
    for (int r = 0; r < 8; ++r) {
        float m = s[0][r];
        #pragma unroll
        for (int f = 1; f < 4; ++f) m = fmaxf(m, s[f][r]);
        for (int d = 1; d < 16; d <<= 1) m = fmaxf(m, __shfl_xor(m, d, 32));
        float sum = 0.f;
        #pragma unroll
        for (int f = 0; f < 4; ++f) {
            float e = __expf(s[f][r] - m);
            s[f][r] = e;
            sum += e;
        }
        for (int d = 1; d < 16; d <<= 1) sum += __shfl_xor(sum, d, 32);
        float inv = 1.0f / sum;
        int grow = rowBase + r + 8 * half;
        ll outBase = ((ll)z * NN + grow) * 64;
        #pragma unroll
        for (int f = 0; f < 4; ++f)
            attn2[outBase + 16 * f + lr] = f2bf(s[f][r] * inv);
    }
}

// ---------------------------------------------------------------------------
// Host side
// ---------------------------------------------------------------------------
static void launch_gemm(hipStream_t st, const u16* A, const u16* B, void* C,
                        const float* bias, int M, int Nn, int K, int batches, int div,
                        ll aO, ll aI, ll aRow, ll bO, ll bI, ll bN, ll bK,
                        ll cO, ll cI, ll cM, ll cN, float scale, int outBF16) {
    dim3 grid((M + 127) / 128, (Nn + 63) / 64, batches);
    k_gemm_bf16<<<grid, 256, 0, st>>>(A, B, C, bias, M, Nn, K, div,
                                      aO, aI, aRow, bO, bI, bN, bK,
                                      cO, cI, cM, cN, scale, outBF16);
}

extern "C" void kernel_launch(void* const* d_in, const int* in_sizes, int n_in,
                              void* d_out, int out_size, void* d_ws, size_t ws_size,
                              hipStream_t stream) {
    (void)in_sizes; (void)n_in; (void)out_size; (void)ws_size;

    const float* x     = (const float*)d_in[0];
    const float* y     = (const float*)d_in[1];
    const float* ln1w  = (const float*)d_in[2];
    const float* ln1b  = (const float*)d_in[3];
    const float* ln2w  = (const float*)d_in[4];
    const float* ln2b  = (const float*)d_in[5];
    const float* wq1   = (const float*)d_in[6];
    const float* wq2   = (const float*)d_in[7];
    const float* wkv   = (const float*)d_in[8];
    const float* wproj = (const float*)d_in[9];
    const float* bproj = (const float*)d_in[10];

    char* ws = (char*)d_ws;
    size_t off = 0;
    auto alloc = [&](size_t bytes) { size_t r = off; off += (bytes + 255) & ~(size_t)255; return r; };

    size_t o_wq1b  = alloc((size_t)CC * CC * 2);
    size_t o_wq2b  = alloc((size_t)CC * CC * 2);
    size_t o_wkvb  = alloc((size_t)2 * CC * CC * 2);
    size_t o_wprjb = alloc((size_t)CC * CC * 2);
    size_t o_xln   = alloc((size_t)ROWS * CC * 2);
    size_t o_ymu   = alloc((size_t)ROWS * 4);
    size_t o_yrs   = alloc((size_t)ROWS * 4);
    size_t o_ypool = alloc((size_t)BB * AG * CC * 2);
    size_t o_q     = alloc((size_t)ROWS * CC * 2);
    size_t o_kv    = alloc((size_t)ROWS * 2 * CC * 2);
    size_t o_at    = alloc((size_t)BB * AG * CC * 2);
    size_t o_s1    = alloc((size_t)BH * AG * NN * 4);
    size_t o_attn1 = alloc((size_t)BH * AG * NN * 2);
    size_t o_agv   = alloc((size_t)BH * AG * HDIM * 2);
    size_t o_attn2 = alloc((size_t)BH * NN * 64 * 2);
    size_t o_outh  = alloc((size_t)ROWS * CC * 2);

    u16* wq1b  = (u16*)(ws + o_wq1b);
    u16* wq2b  = (u16*)(ws + o_wq2b);
    u16* wkvb  = (u16*)(ws + o_wkvb);
    u16* wprjb = (u16*)(ws + o_wprjb);
    u16* xln   = (u16*)(ws + o_xln);
    float* ymu = (float*)(ws + o_ymu);
    float* yrs = (float*)(ws + o_yrs);
    u16* ypool = (u16*)(ws + o_ypool);
    u16* qbuf  = (u16*)(ws + o_q);
    u16* kvbuf = (u16*)(ws + o_kv);
    u16* atbuf = (u16*)(ws + o_at);
    float* s1  = (float*)(ws + o_s1);
    u16* attn1 = (u16*)(ws + o_attn1);
    u16* agv   = (u16*)(ws + o_agv);
    u16* attn2 = (u16*)(ws + o_attn2);
    u16* outh  = (u16*)(ws + o_outh);

    // 1) weights -> bf16
    k_f32_to_bf16<<<(CC * CC + 255) / 256, 256, 0, stream>>>(wq1, wq1b, CC * CC);
    k_f32_to_bf16<<<(CC * CC + 255) / 256, 256, 0, stream>>>(wq2, wq2b, CC * CC);
    k_f32_to_bf16<<<(2 * CC * CC + 255) / 256, 256, 0, stream>>>(wkv, wkvb, 2 * CC * CC);
    k_f32_to_bf16<<<(CC * CC + 255) / 256, 256, 0, stream>>>(wproj, wprjb, CC * CC);

    // 2) LN(x) -> bf16, LN-stats(y), pooled LN(y) -> agent inputs
    k_ln_rows<<<ROWS / 256, 256, 0, stream>>>(x, ln1w, ln1b, xln);
    k_ln_stats<<<ROWS / 256, 256, 0, stream>>>(y, ymu, yrs);
    k_pool_ln<<<BB * AG, 256, 0, stream>>>(y, ymu, yrs, ln2w, ln2b, ypool);

    // 3) q = xln @ wq1^T  (bf16 out)
    launch_gemm(stream, xln, wq1b, qbuf, nullptr, ROWS, CC, CC, 1, 1,
                0, 0, CC, 0, 0, CC, 1, 0, 0, CC, 1, 1.0f, 1);
    // 4) kv = xln @ wkv^T
    launch_gemm(stream, xln, wkvb, kvbuf, nullptr, ROWS, 2 * CC, CC, 1, 1,
                0, 0, CC, 0, 0, CC, 1, 0, 0, 2 * CC, 1, 1.0f, 1);
    // 5) agent_tokens = ypool @ wq2^T  (pool commutes with linear proj)
    launch_gemm(stream, ypool, wq2b, atbuf, nullptr, BB * AG, CC, CC, 1, 1,
                0, 0, CC, 0, 0, CC, 1, 0, 0, CC, 1, 1.0f, 1);

    // 6) agent scores: s1[bh,a,n] = 0.125 * at . k   (fp32 out)
    launch_gemm(stream, atbuf, kvbuf, s1, nullptr, AG, NN, HDIM, BH, NHD,
                (ll)AG * CC, HDIM, CC,
                (ll)NN * 2 * CC, HDIM, 2 * CC, 1,
                (ll)NHD * AG * NN, (ll)AG * NN, NN, 1,
                0.125f, 0);
    // 7) softmax over n -> bf16 attn1
    k_softmax3136<<<BH * AG, 256, 0, stream>>>(s1, attn1);
    // 8) agent_v[bh,a,d] = attn1 @ v   (B operand = v^T via strides)
    launch_gemm(stream, attn1, kvbuf + CC, agv, nullptr, AG, HDIM, NN, BH, NHD,
                (ll)NHD * AG * NN, (ll)AG * NN, NN,
                (ll)NN * 2 * CC, HDIM, 1, 2 * CC,
                (ll)NHD * AG * HDIM, (ll)AG * HDIM, HDIM, 1,
                1.0f, 1);
    // 9) fused q_attn softmax -> bf16 attn2 [96][3136][64] (cols >=49 are 0)
    k_qattn<<<dim3(NN / 64, BH), 128, 0, stream>>>(qbuf, atbuf, attn2);
    // 10) out_heads[b,n,h*64+d] = attn2 @ agent_v
    launch_gemm(stream, attn2, agv, outh, nullptr, NN, HDIM, AG, BH, NHD,
                (ll)NHD * NN * 64, (ll)NN * 64, 64,
                (ll)NHD * AG * HDIM, (ll)AG * HDIM, 1, HDIM,
                (ll)NN * CC, HDIM, CC, 1,
                1.0f, 1);
    // 11) final projection + bias, stored directly as (B, C, H, W) fp32
    launch_gemm(stream, outh, wprjb, d_out, bproj, NN, CC, CC, BB, 1,
                (ll)NN * CC, 0, CC,
                0, 0, CC, 1,
                (ll)CC * NN, 0, 1, NN,
                1.0f, 0);
}
